// HybridTwoTower_65506841198666
// MI455X (gfx1250) — matrix operand order repaired and checked by reading, not compiled, not run on hardware
//
#include <hip/hip_runtime.h>
#include <hip/hip_bf16.h>
#include <math.h>

#define NUM_ITEMS  100000
#define NUM_USERS  16384
#define RECENT_K   50

typedef __attribute__((ext_vector_type(16))) __bf16 v16bf;
typedef __attribute__((ext_vector_type(8)))  __bf16 v8bf;
typedef __attribute__((ext_vector_type(8)))  float  v8f;

// ---------------------------------------------------------------------------
// Fragment loaders for CDNA5 WMMA 16x16x32 bf16: contiguous k = 16*hi + j in both A and B
// ---------------------------------------------------------------------------

// A (16x32, M x K), row-major tile with leading dim `ld` (elements).
// lane m = L&15, hi = L>>4; frag[j] <- K = hi*16 + j  (j=0..15)
__device__ __forceinline__ v16bf load_a_frag(const __bf16* tile, int ld, int kbase, int lane) {
  const int m  = lane & 15;
  const int hi = lane >> 4;
  const __bf16* p = tile + m * ld + kbase + hi * 16;
  v8bf lo = *(const v8bf*)(p);
  v8bf hv = *(const v8bf*)(p + 8);
  v16bf r;
#pragma unroll
  for (int i = 0; i < 8; ++i) { r[i] = lo[i]; r[i + 8] = hv[i]; }
  return r;
}

// B (32x16, K x N) loaded from a PRE-TRANSPOSED weight Bt[N][K] (row-major, ld = K).
// lane n = L&15, hk = L>>4; frag[j] <- K = hk*16 + j  => 16 contiguous bf16.
__device__ __forceinline__ v16bf load_b_frag(const __bf16* Bt, int K, int nbase, int kbase, int lane) {
  const int n  = lane & 15;
  const int hk = lane >> 4;
  const __bf16* p = Bt + (size_t)(nbase + n) * K + kbase + hk * 16;
  v8bf lo = *(const v8bf*)(p);
  v8bf hv = *(const v8bf*)(p + 8);
  v16bf r;
#pragma unroll
  for (int i = 0; i < 8; ++i) { r[i] = lo[i]; r[i + 8] = hv[i]; }
  return r;
}

__device__ __forceinline__ v8f wmma_bf16(v16bf a, v16bf b, v8f c) {
  return __builtin_amdgcn_wmma_f32_16x16x32_bf16(false, a, false, b, (short)0, c, false, false);
}

// ---------------------------------------------------------------------------
// Weight conversion: fp32 [K][N] -> bf16 [N][K] (transposed for B fragments)
// ---------------------------------------------------------------------------
__global__ void cvt_transpose_kernel(const float* __restrict__ src, __bf16* __restrict__ dst,
                                     int K, int N) {
  int t = blockIdx.x * blockDim.x + threadIdx.x;
  if (t < K * N) {
    int n = t / K;
    int k = t - n * K;
    dst[t] = (__bf16)src[(size_t)k * N + n];
  }
}

__global__ void cvt_copy_kernel(const float* __restrict__ src, __bf16* __restrict__ dst, int count) {
  int t = blockIdx.x * blockDim.x + threadIdx.x;
  if (t < count) dst[t] = (__bf16)src[t];
}

// ---------------------------------------------------------------------------
// Item tower: one wave per 32 items (two 16-row sub-tiles share each B frag).
//   geo = relu(feat@Wg+bg); xi=[geo, gcn_item]; h1=relu(xi@Wi1+bi1);
//   item_emb = normalize(h1@Wi2+bi2)  -> out rows [0, NUM_ITEMS)
// ---------------------------------------------------------------------------
__global__ __launch_bounds__(32) void item_tower_kernel(
    const float* __restrict__ item_feat, const float* __restrict__ gcn_item,
    const float* __restrict__ Wg, const float* __restrict__ bg,
    const __bf16* __restrict__ Wi1T, const float* __restrict__ bi1,
    const __bf16* __restrict__ Wi2T, const float* __restrict__ bi2,
    float* __restrict__ out) {
  __shared__ __align__(16) __bf16 s_xi[32 * 160];
  __shared__ __align__(16) __bf16 s_h1[32 * 256];
  __shared__ __align__(16) float  s_o [32 * 128];
  __shared__ float s_scale[32];

  const int lane = threadIdx.x;
  const int i0   = blockIdx.x * 32;

  // Build xi tile: 32 geo cols (lane == col) + 128 gcn cols (lane*4 vectorized)
  {
    const float wg0 = Wg[lane];
    const float wg1 = Wg[32 + lane];
    const float b   = bg[lane];
    for (int r = 0; r < 32; ++r) {
      const float f0 = item_feat[(size_t)(i0 + r) * 2 + 0];
      const float f1 = item_feat[(size_t)(i0 + r) * 2 + 1];
      s_xi[r * 160 + lane] = (__bf16)fmaxf(f0 * wg0 + f1 * wg1 + b, 0.f);
      const float4 e = *(const float4*)(gcn_item + (size_t)(i0 + r) * 128 + lane * 4);
      __bf16* d = s_xi + r * 160 + 32 + lane * 4;
      d[0] = (__bf16)e.x; d[1] = (__bf16)e.y; d[2] = (__bf16)e.z; d[3] = (__bf16)e.w;
    }
  }
  __syncthreads();

  const int nsel = lane & 15;
  const int m0   = (lane >> 4) * 8;

  // h1 = relu(xi @ Wi1 + bi1): [32,256], K=160
  for (int nt = 0; nt < 16; ++nt) {
    const int n = nt * 16 + nsel;
    const float b = bi1[n];
    v8f acc0, acc1;
#pragma unroll
    for (int r = 0; r < 8; ++r) { acc0[r] = b; acc1[r] = b; }
#pragma unroll
    for (int kt = 0; kt < 5; ++kt) {
      v16bf w  = load_b_frag(Wi1T, 160, nt * 16, kt * 32, lane);
      v16bf a0 = load_a_frag(s_xi,            160, kt * 32, lane);
      v16bf a1 = load_a_frag(s_xi + 16 * 160, 160, kt * 32, lane);
      acc0 = wmma_bf16(a0, w, acc0);
      acc1 = wmma_bf16(a1, w, acc1);
    }
#pragma unroll
    for (int r = 0; r < 8; ++r) {
      s_h1[(m0 + r) * 256 + n]      = (__bf16)fmaxf(acc0[r], 0.f);
      s_h1[(16 + m0 + r) * 256 + n] = (__bf16)fmaxf(acc1[r], 0.f);
    }
  }
  __syncthreads();

  // o = h1 @ Wi2 + bi2: [32,128], K=256
  for (int nt = 0; nt < 8; ++nt) {
    const int n = nt * 16 + nsel;
    const float b = bi2[n];
    v8f acc0, acc1;
#pragma unroll
    for (int r = 0; r < 8; ++r) { acc0[r] = b; acc1[r] = b; }
#pragma unroll
    for (int kt = 0; kt < 8; ++kt) {
      v16bf w  = load_b_frag(Wi2T, 256, nt * 16, kt * 32, lane);
      v16bf a0 = load_a_frag(s_h1,            256, kt * 32, lane);
      v16bf a1 = load_a_frag(s_h1 + 16 * 256, 256, kt * 32, lane);
      acc0 = wmma_bf16(a0, w, acc0);
      acc1 = wmma_bf16(a1, w, acc1);
    }
#pragma unroll
    for (int r = 0; r < 8; ++r) {
      s_o[(m0 + r) * 128 + n]      = acc0[r];
      s_o[(16 + m0 + r) * 128 + n] = acc1[r];
    }
  }
  __syncthreads();

  {
    float ss = 0.f;
    for (int c = 0; c < 128; ++c) { float v = s_o[lane * 128 + c]; ss += v * v; }
    s_scale[lane] = 1.f / fmaxf(sqrtf(ss), 1e-12f);
  }
  __syncthreads();

  for (int r = 0; r < 32; ++r) {
    float4 v = *(const float4*)(s_o + r * 128 + lane * 4);
    const float sc = s_scale[r];
    v.x *= sc; v.y *= sc; v.z *= sc; v.w *= sc;
    *(float4*)(out + (size_t)(i0 + r) * 128 + lane * 4) = v;
  }
}

// ---------------------------------------------------------------------------
// User tower: one wave per 32 users.  Algebraic rewrite:
//   scores[u,j] = h[u,j] . (q[u] @ Wk^T) + q[u].bk    (removes [U*K,128]x[128,128] GEMM)
// ---------------------------------------------------------------------------
__global__ __launch_bounds__(32) void user_tower_kernel(
    const float* __restrict__ gcn_user, const int* __restrict__ user_items,
    const __bf16* __restrict__ WqT, const float* __restrict__ bq,
    const __bf16* __restrict__ WkB, const float* __restrict__ bk,
    const __bf16* __restrict__ Wu1T, const float* __restrict__ bu1,
    const __bf16* __restrict__ Wu2T, const float* __restrict__ bu2,
    float* __restrict__ out) {
  __shared__ __align__(16) __bf16 s_fused[32 * 256];   // [gcn_user | rep] in bf16
  __shared__ __align__(16) __bf16 s_qbf [32 * 128];
  __shared__ __align__(16) float  s_qk  [32 * 128];
  __shared__ __align__(16) int    s_idx [32 * RECENT_K];
  __shared__ __align__(16) float  s_sc  [32 * RECENT_K];
  __shared__ __align__(16) __bf16 s_h2  [32 * 256];
  __shared__ __align__(16) float  s_o   [32 * 128];
  __shared__ float s_qb[32];
  __shared__ float s_scale[32];

  const int lane = threadIdx.x;
  const int u0   = blockIdx.x * 32;
  const float* item_emb = out;  // item rows already written by item_tower_kernel

  for (int r = 0; r < 32; ++r) {
    const float4 g = *(const float4*)(gcn_user + (size_t)(u0 + r) * 128 + lane * 4);
    __bf16* d = s_fused + r * 256 + lane * 4;
    d[0] = (__bf16)g.x; d[1] = (__bf16)g.y; d[2] = (__bf16)g.z; d[3] = (__bf16)g.w;
  }
  for (int t = lane; t < 32 * RECENT_K; t += 32)
    s_idx[t] = user_items[(size_t)u0 * RECENT_K + t];
  __syncthreads();

  const int nsel = lane & 15;
  const int m0   = (lane >> 4) * 8;

  // q = gcn_user @ Wq + bq  (A = fused cols 0..127, ld 256)
  for (int nt = 0; nt < 8; ++nt) {
    const int n = nt * 16 + nsel;
    const float b = bq[n];
    v8f acc0, acc1;
#pragma unroll
    for (int r = 0; r < 8; ++r) { acc0[r] = b; acc1[r] = b; }
#pragma unroll
    for (int kt = 0; kt < 4; ++kt) {
      v16bf w  = load_b_frag(WqT, 128, nt * 16, kt * 32, lane);
      v16bf a0 = load_a_frag(s_fused,            256, kt * 32, lane);
      v16bf a1 = load_a_frag(s_fused + 16 * 256, 256, kt * 32, lane);
      acc0 = wmma_bf16(a0, w, acc0);
      acc1 = wmma_bf16(a1, w, acc1);
    }
#pragma unroll
    for (int r = 0; r < 8; ++r) {
      s_qbf[(m0 + r) * 128 + n]      = (__bf16)acc0[r];
      s_qbf[(16 + m0 + r) * 128 + n] = (__bf16)acc1[r];
    }
  }
  __syncthreads();

  {  // qb[u] = q[u].bk  (one user row per lane)
    float s = 0.f;
    for (int c = 0; c < 128; ++c) s += (float)s_qbf[lane * 128 + c] * bk[c];
    s_qb[lane] = s;
  }

  // qk = q @ Wk^T  (Bt[n][k] = Wk[n][k] -> plain bf16 copy of Wk)
  for (int nt = 0; nt < 8; ++nt) {
    const int n = nt * 16 + nsel;
    v8f acc0, acc1;
#pragma unroll
    for (int r = 0; r < 8; ++r) { acc0[r] = 0.f; acc1[r] = 0.f; }
#pragma unroll
    for (int kt = 0; kt < 4; ++kt) {
      v16bf w  = load_b_frag(WkB, 128, nt * 16, kt * 32, lane);
      v16bf a0 = load_a_frag(s_qbf,            128, kt * 32, lane);
      v16bf a1 = load_a_frag(s_qbf + 16 * 128, 128, kt * 32, lane);
      acc0 = wmma_bf16(a0, w, acc0);
      acc1 = wmma_bf16(a1, w, acc1);
    }
#pragma unroll
    for (int r = 0; r < 8; ++r) {
      s_qk[(m0 + r) * 128 + n]      = acc0[r];
      s_qk[(16 + m0 + r) * 128 + n] = acc1[r];
    }
  }
  __syncthreads();

  // scores: L2-resident gather of item_emb rows + dot with qk
  for (int u = 0; u < 32; ++u) {
    const float4* qp = (const float4*)(s_qk + u * 128);
    for (int half = 0; half < 2; ++half) {
      const int j = half * 32 + lane;
      if (j < RECENT_K) {
        const int it = s_idx[u * RECENT_K + j];
        const float4* hp = (const float4*)(item_emb + (size_t)it * 128);
        float s = 0.f;
#pragma unroll 8
        for (int c = 0; c < 32; ++c) {
          float4 h4 = hp[c], q4 = qp[c];
          s += h4.x * q4.x + h4.y * q4.y + h4.z * q4.z + h4.w * q4.w;
        }
        s_sc[u * RECENT_K + j] = s + s_qb[u];
      }
    }
  }
  __syncthreads();

  // softmax over K=50, one user row per lane
  {
    float* row = s_sc + lane * RECENT_K;
    float m = row[0];
    for (int j = 1; j < RECENT_K; ++j) m = fmaxf(m, row[j]);
    float sum = 0.f;
    for (int j = 0; j < RECENT_K; ++j) { float e = __expf(row[j] - m); row[j] = e; sum += e; }
    const float inv = 1.f / sum;
    for (int j = 0; j < RECENT_K; ++j) row[j] *= inv;
  }
  __syncthreads();

  // rep[u] = sum_j alpha * item_emb[idx]  -> fused cols 128..255 (bf16)
  {
    const int c = lane * 4;
    for (int u = 0; u < 32; ++u) {
      float ax = 0.f, ay = 0.f, az = 0.f, aw = 0.f;
      for (int j = 0; j < RECENT_K; ++j) {
        const float a  = s_sc[u * RECENT_K + j];
        const int   it = s_idx[u * RECENT_K + j];
        const float4 h4 = *(const float4*)(item_emb + (size_t)it * 128 + c);
        ax += a * h4.x; ay += a * h4.y; az += a * h4.z; aw += a * h4.w;
      }
      __bf16* d = s_fused + u * 256 + 128 + c;
      d[0] = (__bf16)ax; d[1] = (__bf16)ay; d[2] = (__bf16)az; d[3] = (__bf16)aw;
    }
  }
  __syncthreads();

  // u1 = relu(fused @ Wu1 + bu1): [32,256], K=256
  for (int nt = 0; nt < 16; ++nt) {
    const int n = nt * 16 + nsel;
    const float b = bu1[n];
    v8f acc0, acc1;
#pragma unroll
    for (int r = 0; r < 8; ++r) { acc0[r] = b; acc1[r] = b; }
#pragma unroll
    for (int kt = 0; kt < 8; ++kt) {
      v16bf w  = load_b_frag(Wu1T, 256, nt * 16, kt * 32, lane);
      v16bf a0 = load_a_frag(s_fused,            256, kt * 32, lane);
      v16bf a1 = load_a_frag(s_fused + 16 * 256, 256, kt * 32, lane);
      acc0 = wmma_bf16(a0, w, acc0);
      acc1 = wmma_bf16(a1, w, acc1);
    }
#pragma unroll
    for (int r = 0; r < 8; ++r) {
      s_h2[(m0 + r) * 256 + n]      = (__bf16)fmaxf(acc0[r], 0.f);
      s_h2[(16 + m0 + r) * 256 + n] = (__bf16)fmaxf(acc1[r], 0.f);
    }
  }
  __syncthreads();

  // o = u1 @ Wu2 + bu2: [32,128], K=256
  for (int nt = 0; nt < 8; ++nt) {
    const int n = nt * 16 + nsel;
    const float b = bu2[n];
    v8f acc0, acc1;
#pragma unroll
    for (int r = 0; r < 8; ++r) { acc0[r] = b; acc1[r] = b; }
#pragma unroll
    for (int kt = 0; kt < 8; ++kt) {
      v16bf w  = load_b_frag(Wu2T, 256, nt * 16, kt * 32, lane);
      v16bf a0 = load_a_frag(s_h2,            256, kt * 32, lane);
      v16bf a1 = load_a_frag(s_h2 + 16 * 256, 256, kt * 32, lane);
      acc0 = wmma_bf16(a0, w, acc0);
      acc1 = wmma_bf16(a1, w, acc1);
    }
#pragma unroll
    for (int r = 0; r < 8; ++r) {
      s_o[(m0 + r) * 128 + n]      = acc0[r];
      s_o[(16 + m0 + r) * 128 + n] = acc1[r];
    }
  }
  __syncthreads();

  {
    float ss = 0.f;
    for (int c = 0; c < 128; ++c) { float v = s_o[lane * 128 + c]; ss += v * v; }
    s_scale[lane] = 1.f / fmaxf(sqrtf(ss), 1e-12f);
  }
  __syncthreads();

  for (int r = 0; r < 32; ++r) {
    float4 v = *(const float4*)(s_o + r * 128 + lane * 4);
    const float sc = s_scale[r];
    v.x *= sc; v.y *= sc; v.z *= sc; v.w *= sc;
    *(float4*)(out + (size_t)(NUM_ITEMS + u0 + r) * 128 + lane * 4) = v;
  }
}

// ---------------------------------------------------------------------------
extern "C" void kernel_launch(void* const* d_in, const int* in_sizes, int n_in,
                              void* d_out, int out_size, void* d_ws, size_t ws_size,
                              hipStream_t stream) {
  (void)in_sizes; (void)n_in; (void)out_size; (void)ws_size;
  const float* item_feat  = (const float*)d_in[0];
  const float* gcn_item   = (const float*)d_in[1];
  const float* gcn_user   = (const float*)d_in[2];
  const int*   user_items = (const int*)  d_in[3];
  const float* Wg  = (const float*)d_in[4];
  const float* bg  = (const float*)d_in[5];
  const float* Wi1 = (const float*)d_in[6];
  const float* bi1 = (const float*)d_in[7];
  const float* Wi2 = (const float*)d_in[8];
  const float* bi2 = (const float*)d_in[9];
  const float* Wq  = (const float*)d_in[10];
  const float* bq  = (const float*)d_in[11];
  const float* Wk  = (const float*)d_in[12];
  const float* bk  = (const float*)d_in[13];
  const float* Wu1 = (const float*)d_in[14];
  const float* bu1 = (const float*)d_in[15];
  const float* Wu2 = (const float*)d_in[16];
  const float* bu2 = (const float*)d_in[17];
  float* out = (float*)d_out;

  // bf16 weight workspace (transposed to [N][K] for contiguous B fragments)
  char* ws = (char*)d_ws;
  __bf16* Wi1T = (__bf16*)(ws + 0);        // [256][160]  81920 B
  __bf16* Wi2T = (__bf16*)(ws + 81920);    // [128][256]  65536 B
  __bf16* WqT  = (__bf16*)(ws + 147456);   // [128][128]  32768 B
  __bf16* WkB  = (__bf16*)(ws + 180224);   // [128][128]  32768 B (Wk as-is == (Wk^T)^T)
  __bf16* Wu1T = (__bf16*)(ws + 212992);   // [256][256] 131072 B
  __bf16* Wu2T = (__bf16*)(ws + 344064);   // [128][256]  65536 B

  auto nb = [](int n) { return (n + 255) / 256; };
  cvt_transpose_kernel<<<nb(160 * 256), 256, 0, stream>>>(Wi1, Wi1T, 160, 256);
  cvt_transpose_kernel<<<nb(256 * 128), 256, 0, stream>>>(Wi2, Wi2T, 256, 128);
  cvt_transpose_kernel<<<nb(128 * 128), 256, 0, stream>>>(Wq,  WqT,  128, 128);
  cvt_copy_kernel     <<<nb(128 * 128), 256, 0, stream>>>(Wk,  WkB,  128 * 128);
  cvt_transpose_kernel<<<nb(256 * 256), 256, 0, stream>>>(Wu1, Wu1T, 256, 256);
  cvt_transpose_kernel<<<nb(256 * 128), 256, 0, stream>>>(Wu2, Wu2T, 256, 128);

  item_tower_kernel<<<NUM_ITEMS / 32, 32, 0, stream>>>(
      item_feat, gcn_item, Wg, bg, Wi1T, bi1, Wi2T, bi2, out);

  user_tower_kernel<<<NUM_USERS / 32, 32, 0, stream>>>(
      gcn_user, user_items, WqT, bq, WkB, bk, Wu1T, bu1, Wu2T, bu2, out);
}
